// Attention_49478023250085
// MI455X (gfx1250) — matrix operand order repaired and checked by reading, not compiled, not run on hardware
//
#include <hip/hip_runtime.h>
#include <cstdint>

// ---------------------------------------------------------------------------
// MI455X (gfx1250) fused MHA, bf16 WMMA path (v_wmma_f32_16x16x32_bf16) with
// async-to-LDS weight staging (GLOBAL_LOAD_ASYNC_TO_LDS_B128 / ASYNCcnt),
// double-buffered 64x64 K-tiles (8 WMMA per barrier).
// B=2, S=2048, D=1024, H=16, Dh=64.  Compute-bound -> everything on WMMA.
// ---------------------------------------------------------------------------

#define DEV static __device__ __forceinline__

typedef __bf16 bf16;
typedef __attribute__((ext_vector_type(16))) __bf16 v16bf;
typedef __attribute__((ext_vector_type(8)))  __bf16 v8bf;
typedef __attribute__((ext_vector_type(8)))  float  v8f;

constexpr int BB = 2;
constexpr int SS = 2048;
constexpr int DD = 1024;
constexpr int HH = 16;
constexpr int DH = 64;
constexpr int ROWS = BB * SS;          // 4096 flat (b,s) rows
constexpr float SCORE_SCALE = 0.125f;  // 1/sqrt(Dh)
constexpr float NEG_BIG = -1e9f;

constexpr int TSTRIDE = 72;            // LDS tile row stride (bf16): 144B, 16B
                                       // aligned, 36-bank step -> conflict-free

DEV v8f vzero() {
  v8f z;
#pragma unroll
  for (int i = 0; i < 8; ++i) z[i] = 0.0f;
  return z;
}

DEV v16bf pack16(v8bf lo, v8bf hi) {
  v16bf r;
#pragma unroll
  for (int i = 0; i < 8; ++i) { r[i] = lo[i]; r[i + 8] = hi[i]; }
  return r;
}

// A-fragment (16x32 bf16): lane holds row (lane&15); 16 contiguous K values
// starting at K = half*16, half = lane>>4 (matches B-fragment K order).
DEV v16bf load_a_frag(const bf16* rowptr, int half) {
  const v8bf* p0 = reinterpret_cast<const v8bf*>(rowptr + half * 16);
  const v8bf* p1 = reinterpret_cast<const v8bf*>(rowptr + half * 16 + 8);
  return pack16(*p0, *p1);
}

// B-fragment (32x16 bf16): lane holds col (lane&15); 16 contiguous K values
// starting at K = half*16.  colptr = &B[col][k0 + half*16], K-contiguous.
DEV v16bf load_b_frag(const bf16* colptr) {
  const v8bf* p = reinterpret_cast<const v8bf*>(colptr);
  return pack16(p[0], p[1]);
}

DEV v8f wmma_bf16(v16bf a, v16bf b, v8f c) {
  return __builtin_amdgcn_wmma_f32_16x16x32_bf16(
      false, a, false, b, (short)0, c, false, false);
}

// Async copy 16B global -> LDS (ASYNCcnt); ISA §15.18 VGLOBAL op 98.
DEV void async_g2l_b128(unsigned lds_addr, const bf16* gaddr) {
  asm volatile("global_load_async_to_lds_b128 %0, %1, off"
               :: "v"(lds_addr), "v"(gaddr) : "memory");
}
DEV void wait_async0() { asm volatile("s_wait_asynccnt 0" ::: "memory"); }

// ---------------------------------------------------------------------------
// Prep kernels (bandwidth-trivial; whole problem fits in the 192MB L2)
// ---------------------------------------------------------------------------
__global__ void k_f2bf(const float* __restrict__ in, bf16* __restrict__ out, int n) {
  int i = blockIdx.x * blockDim.x + threadIdx.x;
  if (i < n) out[i] = (bf16)in[i];
}

// W [h][m][d] -> Wt [h][d][m]  (K-contiguous per output column)
__global__ void k_wT(const float* __restrict__ W, bf16* __restrict__ Wt) {
  int i = blockIdx.x * blockDim.x + threadIdx.x;
  const int total = HH * DH * DD;
  if (i >= total) return;
  int h = i / (DH * DD);
  int rem = i % (DH * DD);
  int d = rem / DD;
  int m = rem % DD;
  Wt[i] = (bf16)W[((size_t)h * DD + m) * DH + d];
}

// W_O [h][d][m] -> WoT [m][h*DH+d]  (K contiguous per output column m)
__global__ void k_woT(const float* __restrict__ Wo, bf16* __restrict__ WoT) {
  int i = blockIdx.x * blockDim.x + threadIdx.x;
  const int total = DD * HH * DH;
  if (i >= total) return;
  int m = i / (HH * DH);
  int hd = i % (HH * DH);
  WoT[i] = (bf16)Wo[(size_t)hd * DD + m];
}

__global__ void k_bsum(const float* __restrict__ bO, float* __restrict__ bs) {
  int m = blockIdx.x * blockDim.x + threadIdx.x;
  if (m >= DD) return;
  float s = 0.0f;
  for (int h = 0; h < HH; ++h) s += bO[(size_t)h * DD + m];
  bs[m] = s;
}

// ---------------------------------------------------------------------------
// Stage a 64(col) x 64(K) bf16 tile (K-contiguous source rows) into LDS.
// 512 16B chunks / 128 threads -> 4 async b128 per thread, fully unrolled.
// ---------------------------------------------------------------------------
DEV void issue_tile64(bf16 (*tile)[TSTRIDE], const bf16* wbase, int k0, int tid) {
#pragma unroll
  for (int i = 0; i < 4; ++i) {
    int ch = tid + i * 128;
    int col = ch >> 3, sub = ch & 7;
    const bf16* g = wbase + (size_t)col * DD + k0 + sub * 8;
    unsigned l = (unsigned)(uintptr_t)&tile[col][sub * 8];
    async_g2l_b128(l, g);
  }
}

// ---------------------------------------------------------------------------
// GEMM body shared by k_qkv / k_outproj: 64(M) x 64(N) block, 4 waves, each
// wave a 16x64 strip; weight tile async-staged + double-buffered; inner step
// loads all 4 B-fragments before issuing 4 WMMAs (ILP, batched ds_loads).
// ---------------------------------------------------------------------------
DEV void gemm_64x64(v8f acc[4], const bf16* arow, const bf16* wbase,
                    bf16 (*wtile)[64][TSTRIDE], int tid, int half, int ln) {
  issue_tile64(wtile[0], wbase, 0, tid);
  for (int k0 = 0; k0 < DD; k0 += 64) {
    const int buf = (k0 >> 6) & 1;
    wait_async0();
    __syncthreads();                       // tile[buf] visible to all waves
    if (k0 + 64 < DD) issue_tile64(wtile[buf ^ 1], wbase, k0 + 64, tid);

#pragma unroll
    for (int kk = 0; kk < 64; kk += 32) {
      v16bf a = load_a_frag(arow + k0 + kk, half);
      v16bf bfr[4];
#pragma unroll
      for (int t = 0; t < 4; ++t)
        bfr[t] = load_b_frag(&wtile[buf][t * 16 + ln][kk + half * 16]);
#pragma unroll
      for (int t = 0; t < 4; ++t)
        acc[t] = wmma_bf16(a, bfr[t], acc[t]);
    }
  }
}

// ---------------------------------------------------------------------------
// QKV projection: out[row][h][d] = sum_m xb[row][m] * Wt[h][d][m] + bias[h][d]
// V stored transposed [b][h][d][s] for the PV GEMM.
// ---------------------------------------------------------------------------
__global__ __launch_bounds__(128) void k_qkv(const bf16* __restrict__ xb,
                                             const bf16* __restrict__ Wt,
                                             const float* __restrict__ bias,
                                             bf16* __restrict__ outQK,
                                             bf16* __restrict__ outVT,
                                             int isV) {
  __shared__ __align__(16) bf16 wtile[2][64][TSTRIDE];

  const int tid = threadIdx.x;
  const int lane = tid & 31, wv = tid >> 5, half = lane >> 4, ln = lane & 15;
  const int h = blockIdx.y;
  const int m_base = blockIdx.x * 64 + wv * 16;
  const bf16* wbase = Wt + (size_t)h * DH * DD;  // [col 64][K 1024]

  v8f acc[4];
#pragma unroll
  for (int t = 0; t < 4; ++t) acc[t] = vzero();

  const bf16* arow = xb + (size_t)(m_base + ln) * DD;
  gemm_64x64(acc, arow, wbase, wtile, tid, half, ln);

#pragma unroll
  for (int t = 0; t < 4; ++t) {
#pragma unroll
    for (int r = 0; r < 8; ++r) {
      int M = half * 8 + r;
      int row = m_base + M;
      int d = t * 16 + ln;
      float v = acc[t][r] + bias[h * DH + d];
      if (!isV) {
        outQK[((size_t)row * HH + h) * DH + d] = (bf16)v;
      } else {
        int bidx = row / SS, s = row % SS;
        outVT[(((size_t)bidx * HH + h) * DH + d) * SS + s] = (bf16)v;
      }
    }
  }
}

// ---------------------------------------------------------------------------
// Flash-style causal attention.  Wave owns a 16-row q tile (Q fragments in
// registers); per 32-wide causal k-chunk: batched K-fragment loads -> 4 WMMA
// scores, online softmax (shfl within lane halves = C-layout rows), P routed
// C->A through wave-private LDS, batched V loads -> 4 WMMA into Z.
// ---------------------------------------------------------------------------
__global__ __launch_bounds__(128) void k_attn(const bf16* __restrict__ qb,
                                              const bf16* __restrict__ kb,
                                              const bf16* __restrict__ vT,
                                              bf16* __restrict__ zb) {
  __shared__ __align__(16) bf16 ldsP[4][16][40];

  const int lane = threadIdx.x & 31;
  const int wv = threadIdx.x >> 5;
  const int half = lane >> 4;
  const int ln = lane & 15;
  const int h = blockIdx.y;
  const int b = blockIdx.z;
  const int q0 = blockIdx.x * 64 + wv * 16;
  const int q_hi = q0 + 15;

  v8f accZ[4];
#pragma unroll
  for (int t = 0; t < 4; ++t) accZ[t] = vzero();
  float rowm[8], rowsum[8];
#pragma unroll
  for (int r = 0; r < 8; ++r) { rowm[r] = NEG_BIG; rowsum[r] = 0.0f; }

  // Q A-fragments are causal-loop invariant: hoist into registers.
  const bf16* qrow = qb + ((size_t)(b * SS + q0 + ln) * HH + h) * DH;
  v16bf qfrag[2];
#pragma unroll
  for (int d0 = 0; d0 < DH; d0 += 32) qfrag[d0 >> 5] = load_a_frag(qrow + d0, half);

  const bf16* kbase = kb + (size_t)b * SS * HH * DH + (size_t)h * DH;
  const bf16* vbase = vT + ((size_t)b * HH + h) * DH * SS;

  for (int c0 = 0; c0 <= q_hi; c0 += 32) {
    // ---- batched K-fragment loads, then 4 score WMMAs (K = Dh = 64) ----
    v16bf kfA[2], kfB[2];
#pragma unroll
    for (int d0 = 0; d0 < DH; d0 += 32) {
      kfA[d0 >> 5] = load_b_frag(kbase + (size_t)(c0 + ln) * HH * DH + d0 + half * 16);
      kfB[d0 >> 5] = load_b_frag(kbase + (size_t)(c0 + 16 + ln) * HH * DH + d0 + half * 16);
    }
    v8f sA = vzero(), sB = vzero();
#pragma unroll
    for (int d0 = 0; d0 < DH; d0 += 32) {
      sA = wmma_bf16(qfrag[d0 >> 5], kfA[d0 >> 5], sA);
      sB = wmma_bf16(qfrag[d0 >> 5], kfB[d0 >> 5], sB);
    }

    // ---- causal mask + online softmax ----
    float pA[8], pB[8], corr[8];
#pragma unroll
    for (int r = 0; r < 8; ++r) {
      int qi = q0 + half * 8 + r;
      float av = sA[r] * SCORE_SCALE;
      float bv = sB[r] * SCORE_SCALE;
      if (c0 + ln > qi) av = NEG_BIG;
      if (c0 + 16 + ln > qi) bv = NEG_BIG;
      float mx = fmaxf(av, bv);
#pragma unroll
      for (int off = 1; off < 16; off <<= 1)
        mx = fmaxf(mx, __shfl_xor(mx, off, 16));
      float newm = fmaxf(rowm[r], mx);
      float c = __expf(rowm[r] - newm);
      float ea = __expf(av - newm);
      float eb = __expf(bv - newm);
      float s = ea + eb;
#pragma unroll
      for (int off = 1; off < 16; off <<= 1)
        s += __shfl_xor(s, off, 16);
      rowsum[r] = rowsum[r] * c + s;
      rowm[r] = newm;
      corr[r] = c;
      pA[r] = ea;
      pB[r] = eb;
    }

#pragma unroll
    for (int t = 0; t < 4; ++t)
#pragma unroll
      for (int r = 0; r < 8; ++r) accZ[t][r] = accZ[t][r] * corr[r];

    // ---- P: C-layout regs -> LDS -> A-layout fragment (wave-private) ----
#pragma unroll
    for (int r = 0; r < 8; ++r) {
      int M = half * 8 + r;
      ldsP[wv][M][ln] = (bf16)pA[r];
      ldsP[wv][M][16 + ln] = (bf16)pB[r];
    }
    asm volatile("s_wait_dscnt 0" ::: "memory");

    v16bf pfrag;
    {
      const bf16* prow = &ldsP[wv][ln][0];
      const v8bf* p0 = reinterpret_cast<const v8bf*>(prow + half * 16);
      const v8bf* p1 = reinterpret_cast<const v8bf*>(prow + half * 16 + 8);
      pfrag = pack16(*p0, *p1);
    }

    // ---- batched V loads, then Z += P(16x32) x V(32x64) ----
    v16bf vf[4];
#pragma unroll
    for (int t = 0; t < 4; ++t)
      vf[t] = load_b_frag(vbase + (size_t)(t * 16 + ln) * SS + c0 + half * 16);
#pragma unroll
    for (int t = 0; t < 4; ++t)
      accZ[t] = wmma_bf16(pfrag, vf[t], accZ[t]);
    asm volatile("" ::: "memory");  // keep LDS reads ahead of next-iter writes
  }

  // ---- normalize and emit z as flat [row][h*Dh + d] bf16 ----
#pragma unroll
  for (int t = 0; t < 4; ++t) {
#pragma unroll
    for (int r = 0; r < 8; ++r) {
      int M = half * 8 + r;
      int row = b * SS + q0 + M;
      int d = t * 16 + ln;
      float z = accZ[t][r] / rowsum[r];
      zb[(size_t)row * (HH * DH) + h * DH + d] = (bf16)z;
    }
  }
}

// ---------------------------------------------------------------------------
// Output projection: out[row][n] = sum_k zb[row][k] * WoT[n][k] + bsum[n]
// ---------------------------------------------------------------------------
__global__ __launch_bounds__(128) void k_outproj(const bf16* __restrict__ zb,
                                                 const bf16* __restrict__ WoT,
                                                 const float* __restrict__ bsum,
                                                 float* __restrict__ out) {
  __shared__ __align__(16) bf16 wtile[2][64][TSTRIDE];

  const int tid = threadIdx.x;
  const int lane = tid & 31, wv = tid >> 5, half = lane >> 4, ln = lane & 15;
  const int m_base = blockIdx.x * 64 + wv * 16;
  const int n_base = blockIdx.y * 64;
  const bf16* wbase = WoT + (size_t)n_base * (HH * DH);  // [col 64][K 1024]

  v8f acc[4];
#pragma unroll
  for (int t = 0; t < 4; ++t) acc[t] = vzero();

  const bf16* arow = zb + (size_t)(m_base + ln) * (HH * DH);
  gemm_64x64(acc, arow, wbase, wtile, tid, half, ln);

#pragma unroll
  for (int t = 0; t < 4; ++t) {
#pragma unroll
    for (int r = 0; r < 8; ++r) {
      int M = half * 8 + r;
      int row = m_base + M;
      int n = n_base + t * 16 + ln;
      out[(size_t)row * DD + n] = acc[t][r] + bsum[n];
    }
  }
}

// ---------------------------------------------------------------------------
// Host launcher.  Workspace (~48 MB):
//  xb 8MB | WqT/WkT/WvT 2MB ea | WoT 2MB | bsum 4KB | q 8MB | k 8MB | vT 8MB | z 8MB
// ---------------------------------------------------------------------------
extern "C" void kernel_launch(void* const* d_in, const int* in_sizes, int n_in,
                              void* d_out, int out_size, void* d_ws, size_t ws_size,
                              hipStream_t stream) {
  const float* x  = (const float*)d_in[0];
  const float* WQ = (const float*)d_in[1];
  const float* bQ = (const float*)d_in[2];
  const float* WK = (const float*)d_in[3];
  const float* bK = (const float*)d_in[4];
  const float* WV = (const float*)d_in[5];
  const float* WO = (const float*)d_in[6];
  const float* bV = (const float*)d_in[7];
  const float* bO = (const float*)d_in[8];
  float* out = (float*)d_out;

  char* ws = (char*)d_ws;
  size_t off = 0;
  auto take = [&](size_t bytes) -> void* {
    void* p = ws + off;
    off += (bytes + 255) & ~(size_t)255;
    return p;
  };
  bf16* xb   = (bf16*)take((size_t)ROWS * DD * sizeof(bf16));
  bf16* WqT  = (bf16*)take((size_t)HH * DH * DD * sizeof(bf16));
  bf16* WkT  = (bf16*)take((size_t)HH * DH * DD * sizeof(bf16));
  bf16* WvT  = (bf16*)take((size_t)HH * DH * DD * sizeof(bf16));
  bf16* WoT  = (bf16*)take((size_t)DD * HH * DH * sizeof(bf16));
  float* bsO = (float*)take((size_t)DD * sizeof(float));
  bf16* qb   = (bf16*)take((size_t)ROWS * HH * DH * sizeof(bf16));
  bf16* kbuf = (bf16*)take((size_t)ROWS * HH * DH * sizeof(bf16));
  bf16* vTb  = (bf16*)take((size_t)BB * HH * DH * SS * sizeof(bf16));
  bf16* zb   = (bf16*)take((size_t)ROWS * HH * DH * sizeof(bf16));

  {
    int n = ROWS * DD;
    k_f2bf<<<(n + 255) / 256, 256, 0, stream>>>(x, xb, n);
    int wtot = HH * DH * DD;
    k_wT<<<(wtot + 255) / 256, 256, 0, stream>>>(WQ, WqT);
    k_wT<<<(wtot + 255) / 256, 256, 0, stream>>>(WK, WkT);
    k_wT<<<(wtot + 255) / 256, 256, 0, stream>>>(WV, WvT);
    k_woT<<<(wtot + 255) / 256, 256, 0, stream>>>(WO, WoT);
    k_bsum<<<(DD + 255) / 256, 256, 0, stream>>>(bO, bsO);
  }
  {
    dim3 grid(ROWS / 64, HH);
    k_qkv<<<grid, 128, 0, stream>>>(xb, WqT, bQ, qb,  nullptr, 0);
    k_qkv<<<grid, 128, 0, stream>>>(xb, WkT, bK, kbuf, nullptr, 0);
    k_qkv<<<grid, 128, 0, stream>>>(xb, WvT, bV, nullptr, vTb, 1);
  }
  {
    dim3 grid(SS / 64, HH, BB);
    k_attn<<<grid, 128, 0, stream>>>(qb, kbuf, vTb, zb);
  }
  {
    dim3 grid(ROWS / 64, DD / 64);
    k_outproj<<<grid, 128, 0, stream>>>(zb, WoT, bsO, out);
  }
}
